// RowParallelLinearSwitchMixtureOfExperts_9311489098244
// MI455X (gfx1250) — matrix operand order repaired and checked by reading, not compiled, not run on hardware
//
#include <hip/hip_runtime.h>
#include <stdint.h>

// ---------------- model dims ----------------
constexpr int kD = 1024;   // input dim
constexpr int kH = 4096;   // hidden dim
constexpr int kO = 1024;   // expert output dim
constexpr int kE = 8;      // experts
constexpr int kK = 2;      // top-k
constexpr int kT = 4096;   // tokens (B*S = 2*2048)

// ---------------- GEMM tiling ----------------
constexpr int BM = 64;           // block tile M
constexpr int BN = 128;          // block tile N
constexpr int BK = 64;           // K step (2 WMMA K-steps of 32)
constexpr int LDA = BK + 8;      // LDS A row stride (halves): 72 -> no bank conflicts
constexpr int LDB = BN + 8;      // LDS B row stride (halves): 136 -> no bank conflicts
constexpr int MT = kT / BM;      // max m-tiles per expert

typedef __attribute__((ext_vector_type(16))) __bf16 v16bf;
typedef __attribute__((ext_vector_type(8)))  float  v8f;

union FragAB { uint4 u[2]; v16bf v; };

__device__ __forceinline__ uint16_t f2bf(float f) {
  uint32_t u = __float_as_uint(f);
  u += 0x7FFFu + ((u >> 16) & 1u);   // round-to-nearest-even
  return (uint16_t)(u >> 16);
}

// raw LDS offset (addrspace(3) ptrtoint) for async-to-LDS destinations
__device__ __forceinline__ uint32_t lds_off_of(const void* p) {
  return (uint32_t)(uintptr_t)(__attribute__((address_space(3))) const void*)p;
}

// One tile = 6x GLOBAL_LOAD_ASYNC_TO_LDS_B128 in a single asm block.
// NOTE (ISA ch.10 async pseudocode): the instruction offset is added to BOTH
// the LDS and the global address, so equal-delta chunks share a base pair.
__device__ __forceinline__ void issue6(uint32_t aL, uint32_t bL0, uint32_t bL1,
                                       const uint16_t* aA, const uint16_t* bA0,
                                       const uint16_t* bA1) {
  asm volatile(
      "global_load_async_to_lds_b128 %0, %3, off\n\t"
      "global_load_async_to_lds_b128 %0, %3, off offset:64\n\t"
      "global_load_async_to_lds_b128 %1, %4, off\n\t"
      "global_load_async_to_lds_b128 %1, %4, off offset:16\n\t"
      "global_load_async_to_lds_b128 %2, %5, off\n\t"
      "global_load_async_to_lds_b128 %2, %5, off offset:16\n\t"
      :: "v"(aL), "v"(bL0), "v"(bL1),
         "v"((uint64_t)(uintptr_t)aA),
         "v"((uint64_t)(uintptr_t)bA0),
         "v"((uint64_t)(uintptr_t)bA1)
      : "memory");
}
__device__ __forceinline__ void wait_async_le6() {
  asm volatile("s_wait_asynccnt 0x6" ::: "memory");
}
__device__ __forceinline__ void wait_async_0() {
  asm volatile("s_wait_asynccnt 0x0" ::: "memory");
}

// ---------------------------------------------------------------------------
// 0) zero output + expert counters (ws/out are NOT re-cleared by the harness)
// ---------------------------------------------------------------------------
__global__ void zero_kernel(float* __restrict__ out, int n, int* __restrict__ counts) {
  int i = blockIdx.x * blockDim.x + threadIdx.x;
  if (i < n) out[i] = 0.0f;
  if (i < kE) counts[i] = 0;
}

// ---------------------------------------------------------------------------
// 1) fp32 -> bf16 conversion (vectorized x4)
// ---------------------------------------------------------------------------
__global__ void cvt_bf16_kernel(const float* __restrict__ src,
                                uint16_t* __restrict__ dst, int n4) {
  int i = blockIdx.x * blockDim.x + threadIdx.x;
  if (i >= n4) return;
  float4 f = ((const float4*)src)[i];
  uint32_t lo = (uint32_t)f2bf(f.x) | ((uint32_t)f2bf(f.y) << 16);
  uint32_t hi = (uint32_t)f2bf(f.z) | ((uint32_t)f2bf(f.w) << 16);
  ((uint2*)dst)[i] = make_uint2(lo, hi);
}

// ---------------------------------------------------------------------------
// 2) router: 1 wave32 per token; logits = x@Wr + br; top-2 softmax;
//    atomic append (token, slot, gate) to per-expert lists.
// ---------------------------------------------------------------------------
__global__ void router_kernel(const float* __restrict__ x,
                              const float* __restrict__ Wr,
                              const float* __restrict__ br,
                              int* __restrict__ counts,
                              int* __restrict__ tokL,
                              int* __restrict__ sltL,
                              float* __restrict__ gateL) {
  int wave = threadIdx.x >> 5;
  int lane = threadIdx.x & 31;
  int t = blockIdx.x * 8 + wave;
  if (t >= kT) return;

  float acc[kE] = {};
  for (int d = lane; d < kD; d += 32) {
    float xv = x[(size_t)t * kD + d];
    const float* w = Wr + (size_t)d * kE;
#pragma unroll
    for (int e = 0; e < kE; ++e) acc[e] += xv * w[e];
  }
#pragma unroll
  for (int off = 16; off >= 1; off >>= 1) {
#pragma unroll
    for (int e = 0; e < kE; ++e) acc[e] += __shfl_xor(acc[e], off, 32);
  }
  if (lane == 0) {
#pragma unroll
    for (int e = 0; e < kE; ++e) acc[e] += br[e];
    int i0 = 0; float v0 = acc[0];
    for (int e = 1; e < kE; ++e) if (acc[e] > v0) { v0 = acc[e]; i0 = e; }
    int i1 = (i0 == 0) ? 1 : 0; float v1 = acc[i1];
    for (int e = 0; e < kE; ++e)
      if (e != i0 && acc[e] > v1) { v1 = acc[e]; i1 = e; }
    float e0 = __expf(v0 - v0), e1 = __expf(v1 - v0);
    float inv = 1.0f / (e0 + e1);
    int p0 = atomicAdd(&counts[i0], 1);
    tokL[i0 * kT + p0] = t; sltL[i0 * kT + p0] = t * kK + 0; gateL[i0 * kT + p0] = e0 * inv;
    int p1 = atomicAdd(&counts[i1], 1);
    tokL[i1 * kT + p1] = t; sltL[i1 * kT + p1] = t * kK + 1; gateL[i1 * kT + p1] = e1 * inv;
  }
}

// ---------------------------------------------------------------------------
// GEMM1: h[slot, :] = relu( x[tok, :] @ W1[e] + b1[e] )   (bf16 in, bf16 out)
// grid = (kE * MT, kH / BN), block = 256 (8 waves, 2x4 -> 64x128 tile)
// Double-buffered LDS filled by GLOBAL_LOAD_ASYNC_TO_LDS_B128 (ASYNCcnt).
// ---------------------------------------------------------------------------
__launch_bounds__(256)
__global__ void gemm1_kernel(const uint16_t* __restrict__ xb,
                             const uint16_t* __restrict__ w1b,
                             const float* __restrict__ b1,
                             uint16_t* __restrict__ hb,
                             const int* __restrict__ counts,
                             const int* __restrict__ tokL,
                             const int* __restrict__ sltL) {
  const int e  = blockIdx.x / MT;
  const int mt = blockIdx.x % MT;
  const int rows = counts[e];
  const int m0 = mt * BM;
  if (m0 >= rows) return;
  const int n0 = blockIdx.y * BN;

  __shared__ uint16_t As[2][BM * LDA];
  __shared__ uint16_t Bs[2][BK * LDB];
  __shared__ int toks[BM];
  __shared__ int slts[BM];

  const int tid = threadIdx.x;
  if (tid < BM) {
    int m = m0 + tid;
    int src = (m < rows) ? (e * kT + m) : (e * kT + m0);  // safe fallback row
    toks[tid] = tokL[src];
    slts[tid] = sltL[src];
  }
  __syncthreads();

  const int wave = tid >> 5, lane = tid & 31;
  const int wm = wave & 1, wn = wave >> 1;        // 2 x 4 waves
  const int r = lane & 15, half = lane >> 4;

  v8f c[2][2] = {};

  // per-thread async-copy coordinates
  const int a_row = tid >> 2;                     // 0..63
  const int a_c8  = (tid & 3) * 8;                // halves; +32 second chunk via offset:64
  const int b_k   = tid >> 3;                     // 0..31 (K rows; +32 second chunk)
  const int b_c16 = (tid & 7) * 16;               // halves within BN

  const uint16_t* aA  = xb + (size_t)toks[a_row] * kD + a_c8;
  const uint16_t* bA0 = w1b + (size_t)e * kD * kH + (size_t)b_k * kH + n0 + b_c16;
  const uint16_t* bA1 = bA0 + (size_t)32 * kH;

  uint32_t aL[2], bL0[2], bL1[2];
#pragma unroll
  for (int p = 0; p < 2; ++p) {
    aL[p]  = lds_off_of(&As[p][a_row * LDA + a_c8]);
    bL0[p] = lds_off_of(&Bs[p][b_k * LDB + b_c16]);
    bL1[p] = bL0[p] + 32 * LDB * 2;
  }

  issue6(aL[0], bL0[0], bL1[0], aA, bA0, bA1);
  aA += BK; bA0 += (size_t)BK * kH; bA1 += (size_t)BK * kH;

  int buf = 0;
  for (int k0 = 0; k0 < kD; k0 += BK, buf ^= 1) {
    __syncthreads();                       // prev compute done -> other buffer reusable
    if (k0 + BK < kD) {
      issue6(aL[buf ^ 1], bL0[buf ^ 1], bL1[buf ^ 1], aA, bA0, bA1);
      aA += BK; bA0 += (size_t)BK * kH; bA1 += (size_t)BK * kH;
      wait_async_le6();                    // current tile's 6 DMAs have landed
    } else {
      wait_async_0();
    }
    __syncthreads();                       // all waves' DMAs visible

    const uint16_t* Ab = &As[buf][0];
    const uint16_t* Bb = &Bs[buf][0];
#pragma unroll
    for (int ks = 0; ks < 2; ++ks) {
      FragAB a[2], b[2];
#pragma unroll
      for (int ms = 0; ms < 2; ++ms) {
        int row = wm * 32 + ms * 16 + r;
        a[ms].u[0] = *(const uint4*)(Ab + row * LDA + ks * 32 + half * 8);
        a[ms].u[1] = *(const uint4*)(Ab + row * LDA + ks * 32 + 16 + half * 8);
      }
#pragma unroll
      for (int ns = 0; ns < 2; ++ns) {
        int col = wn * 32 + ns * 16 + r;
#pragma unroll
        for (int s = 0; s < 8; ++s) {
          ((uint16_t*)&b[ns].u[0])[s] = Bb[(ks * 32 + half * 8 + s) * LDB + col];
          ((uint16_t*)&b[ns].u[1])[s] = Bb[(ks * 32 + 16 + half * 8 + s) * LDB + col];
        }
      }
#pragma unroll
      for (int ms = 0; ms < 2; ++ms)
#pragma unroll
        for (int ns = 0; ns < 2; ++ns)
          c[ms][ns] = __builtin_amdgcn_wmma_f32_16x16x32_bf16(
              false, a[ms].v, false, b[ns].v, (short)0, c[ms][ns], false, false);
    }
  }

  // epilogue: +bias, relu, bf16, scatter to h rows by slot
#pragma unroll
  for (int ms = 0; ms < 2; ++ms) {
#pragma unroll
    for (int ns = 0; ns < 2; ++ns) {
      int col = n0 + wn * 32 + ns * 16 + r;
      float bias = b1[(size_t)e * kH + col];
#pragma unroll
      for (int v = 0; v < 8; ++v) {
        int mrow = wm * 32 + ms * 16 + half * 8 + v;
        if (m0 + mrow < rows) {
          float y = c[ms][ns][v] + bias;
          y = y > 0.0f ? y : 0.0f;
          hb[(size_t)slts[mrow] * kH + col] = f2bf(y);
        }
      }
    }
  }
}

// ---------------------------------------------------------------------------
// GEMM2: out[tok, :] += gate * ( h[slot, :] @ W2[e] + b2[e] )
// grid = (kE * MT, kO / BN), block = 256
// ---------------------------------------------------------------------------
__launch_bounds__(256)
__global__ void gemm2_kernel(const uint16_t* __restrict__ hb,
                             const uint16_t* __restrict__ w2b,
                             const float* __restrict__ b2,
                             float* __restrict__ out,
                             const int* __restrict__ counts,
                             const int* __restrict__ tokL,
                             const int* __restrict__ sltL,
                             const float* __restrict__ gateL) {
  const int e  = blockIdx.x / MT;
  const int mt = blockIdx.x % MT;
  const int rows = counts[e];
  const int m0 = mt * BM;
  if (m0 >= rows) return;
  const int n0 = blockIdx.y * BN;

  __shared__ uint16_t As[2][BM * LDA];
  __shared__ uint16_t Bs[2][BK * LDB];
  __shared__ int toks[BM];
  __shared__ int slts[BM];
  __shared__ float gats[BM];

  const int tid = threadIdx.x;
  if (tid < BM) {
    int m = m0 + tid;
    int src = (m < rows) ? (e * kT + m) : (e * kT + m0);
    toks[tid] = tokL[src];
    slts[tid] = sltL[src];
    gats[tid] = gateL[src];
  }
  __syncthreads();

  const int wave = tid >> 5, lane = tid & 31;
  const int wm = wave & 1, wn = wave >> 1;
  const int r = lane & 15, half = lane >> 4;

  v8f c[2][2] = {};

  const int a_row = tid >> 2;
  const int a_c8  = (tid & 3) * 8;
  const int b_k   = tid >> 3;
  const int b_c16 = (tid & 7) * 16;

  const uint16_t* aA  = hb + (size_t)slts[a_row] * kH + a_c8;
  const uint16_t* bA0 = w2b + (size_t)e * kH * kO + (size_t)b_k * kO + n0 + b_c16;
  const uint16_t* bA1 = bA0 + (size_t)32 * kO;

  uint32_t aL[2], bL0[2], bL1[2];
#pragma unroll
  for (int p = 0; p < 2; ++p) {
    aL[p]  = lds_off_of(&As[p][a_row * LDA + a_c8]);
    bL0[p] = lds_off_of(&Bs[p][b_k * LDB + b_c16]);
    bL1[p] = bL0[p] + 32 * LDB * 2;
  }

  issue6(aL[0], bL0[0], bL1[0], aA, bA0, bA1);
  aA += BK; bA0 += (size_t)BK * kO; bA1 += (size_t)BK * kO;

  int buf = 0;
  for (int k0 = 0; k0 < kH; k0 += BK, buf ^= 1) {
    __syncthreads();
    if (k0 + BK < kH) {
      issue6(aL[buf ^ 1], bL0[buf ^ 1], bL1[buf ^ 1], aA, bA0, bA1);
      aA += BK; bA0 += (size_t)BK * kO; bA1 += (size_t)BK * kO;
      wait_async_le6();
    } else {
      wait_async_0();
    }
    __syncthreads();

    const uint16_t* Ab = &As[buf][0];
    const uint16_t* Bb = &Bs[buf][0];
#pragma unroll
    for (int ks = 0; ks < 2; ++ks) {
      FragAB a[2], b[2];
#pragma unroll
      for (int ms = 0; ms < 2; ++ms) {
        int row = wm * 32 + ms * 16 + r;
        a[ms].u[0] = *(const uint4*)(Ab + row * LDA + ks * 32 + half * 8);
        a[ms].u[1] = *(const uint4*)(Ab + row * LDA + ks * 32 + 16 + half * 8);
      }
#pragma unroll
      for (int ns = 0; ns < 2; ++ns) {
        int col = wn * 32 + ns * 16 + r;
#pragma unroll
        for (int s = 0; s < 8; ++s) {
          ((uint16_t*)&b[ns].u[0])[s] = Bb[(ks * 32 + half * 8 + s) * LDB + col];
          ((uint16_t*)&b[ns].u[1])[s] = Bb[(ks * 32 + 16 + half * 8 + s) * LDB + col];
        }
      }
#pragma unroll
      for (int ms = 0; ms < 2; ++ms)
#pragma unroll
        for (int ns = 0; ns < 2; ++ns)
          c[ms][ns] = __builtin_amdgcn_wmma_f32_16x16x32_bf16(
              false, a[ms].v, false, b[ns].v, (short)0, c[ms][ns], false, false);
    }
  }

  // epilogue: +bias, *gate, atomic accumulate into out[token]
#pragma unroll
  for (int ms = 0; ms < 2; ++ms) {
#pragma unroll
    for (int ns = 0; ns < 2; ++ns) {
      int col = n0 + wn * 32 + ns * 16 + r;
      float bias = b2[(size_t)e * kO + col];
#pragma unroll
      for (int v = 0; v < 8; ++v) {
        int mrow = wm * 32 + ms * 16 + half * 8 + v;
        if (m0 + mrow < rows) {
          float y = (c[ms][ns][v] + bias) * gats[mrow];
          __hip_atomic_fetch_add(&out[(size_t)toks[mrow] * kO + col], y,
                                 __ATOMIC_RELAXED, __HIP_MEMORY_SCOPE_AGENT);
        }
      }
    }
  }
}

// ---------------------------------------------------------------------------
extern "C" void kernel_launch(void* const* d_in, const int* in_sizes, int n_in,
                              void* d_out, int out_size, void* d_ws, size_t ws_size,
                              hipStream_t stream) {
  const float* x  = (const float*)d_in[0];
  const float* Wr = (const float*)d_in[1];
  const float* br = (const float*)d_in[2];
  const float* W1 = (const float*)d_in[3];
  const float* b1 = (const float*)d_in[4];
  const float* W2 = (const float*)d_in[5];
  const float* b2 = (const float*)d_in[6];
  float* out = (float*)d_out;

  // workspace layout (bf16 stored as uint16_t)
  uint16_t* xb  = (uint16_t*)d_ws;                       // [kT, kD]
  uint16_t* w1b = xb  + (size_t)kT * kD;                 // [kE, kD, kH]
  uint16_t* w2b = w1b + (size_t)kE * kD * kH;            // [kE, kH, kO]
  uint16_t* hb  = w2b + (size_t)kE * kH * kO;            // [kT*kK, kH]
  int*   tok    = (int*)(hb + (size_t)kT * kK * kH);     // [kE, kT]
  int*   slt    = tok + (size_t)kE * kT;                 // [kE, kT]
  float* gate   = (float*)(slt + (size_t)kE * kT);       // [kE, kT]
  int*   counts = (int*)(gate + (size_t)kE * kT);        // [kE]

  const int nOut = kT * kO;
  zero_kernel<<<(nOut + 255) / 256, 256, 0, stream>>>(out, nOut, counts);

  cvt_bf16_kernel<<<(kT * kD / 4 + 255) / 256, 256, 0, stream>>>(x, xb, kT * kD / 4);
  cvt_bf16_kernel<<<(kE * kD * kH / 4 + 255) / 256, 256, 0, stream>>>(W1, w1b, kE * kD * kH / 4);
  cvt_bf16_kernel<<<(kE * kH * kO / 4 + 255) / 256, 256, 0, stream>>>(W2, w2b, kE * kH * kO / 4);

  router_kernel<<<kT / 8, 256, 0, stream>>>(x, Wr, br, counts, tok, slt, gate);

  gemm1_kernel<<<dim3(kE * MT, kH / BN), 256, 0, stream>>>(xb, w1b, b1, hb, counts, tok, slt);
  gemm2_kernel<<<dim3(kE * MT, kO / BN), 256, 0, stream>>>(hb, w2b, b2, out, counts, tok, slt, gate);
}
